// cRNN_36258113913003
// MI455X (gfx1250) — matrix operand order repaired and checked by reading, not compiled, not run on hardware
//
#include <hip/hip_runtime.h>

// ---------------- problem constants ----------------
#define LAG   168
#define INF_  64          // input feature dim
#define H     1024
#define BATCH 512
#define TT    168
#define K1P   192         // LAG padded up to multiple of 32
#define H3    (3 * H)

// LDS staging for the recurrent GEMM
#define KCH   128               // K-chunk staged per double-buffer slot
#define NCH   (H / KCH)         // 8 chunks
#define APAD  8                 // pad (elements) -> TDM pad: +16B per 256B row
#define AROW  (KCH + APAD)      // 136 bf16 per row (272 B, 17*16B -> aligned)
#define XPAD  8
#define XROW  (INF_ + XPAD)     // 72 bf16 per row (144 B)

typedef __bf16 bf16_t;
typedef __attribute__((ext_vector_type(16))) __bf16 bf16x16;
typedef __attribute__((ext_vector_type(8)))  float  f32x8;
typedef __attribute__((ext_vector_type(4)))  unsigned int u32x4;
typedef __attribute__((ext_vector_type(8)))  int          i32x8;
typedef __attribute__((ext_vector_type(4)))  int          i32x4;

struct FragBF {
    union { bf16x16 v; uint4 q[2]; };
};

// A-matrix 16x32 bf16 fragment (row-major A[M][ldk]):
// lane: m = m0 + (lane&15); kb = k0 + 8*(lane>=16)
// VGPR0-3 = K[kb..kb+7], VGPR4-7 = K[kb+16..kb+23]  -> two contiguous 16B loads
__device__ __forceinline__ void load_fragA(FragBF& f, const bf16_t* __restrict__ base,
                                           int ldk, int m, int kb) {
    const bf16_t* p = base + (size_t)m * ldk + kb;
    f.q[0] = *(const uint4*)(p);
    f.q[1] = *(const uint4*)(p + 16);
}

// B-matrix 32x16 bf16 fragment, weights stored row-major W[N][ldk] (== column-major K x N):
// lane: n = n0 + (lane&15); kb = k0 + 16*(lane>=16); per lane K[k0+8h..+7], K[k0+16+8h..+7] (h = lane>=16)
__device__ __forceinline__ void load_fragB(FragBF& f, const bf16_t* __restrict__ row, int kb) {
    const bf16_t* p = row + (kb & ~16) + ((kb & 16) >> 1);
    f.q[0] = *(const uint4*)(p);
    f.q[1] = *(const uint4*)(p + 16);
}

__device__ __forceinline__ float sigmoid_f(float x) {
    return 1.0f / (1.0f + __expf(-x));
}
__device__ __forceinline__ float tanh_f(float x) {
    float e = __expf(-2.0f * x);
    return (1.0f - e) / (1.0f + e);
}

// ---- Tensor Data Mover: load a 2D bf16 tile [32 x KCH] (row stride H elements)
// from global into LDS at lds_off, with HW padding +16B per 256B row -> AROW layout.
// Descriptor per CDNA5 ISA 8.3/8.4 (group0: count/lds/global/type; group1: dims).
__device__ __forceinline__ void tdm_load_h_tile(const bf16_t* gsrc, unsigned int lds_off) {
    unsigned long long ga = (unsigned long long)(uintptr_t)gsrc;
    u32x4 g0;
    g0[0] = 1u;                                              // count=1 (user D#)
    g0[1] = lds_off;                                         // lds_addr (bytes)
    g0[2] = (unsigned int)ga;                                // global_addr[31:0]
    g0[3] = (unsigned int)((ga >> 32) & 0x01FFFFFFu)         // global_addr[56:32]
          | (2u << 30);                                      // type = 2 ("image")
    i32x8 g1;
    g1[0] = (int)((1u << 16)      // data_size = 1 -> 2 bytes
                | (1u << 20)      // pad_enable
                | (5u << 22)      // pad_interval = 64 DWORDs (256B)
                | (3u << 25));    // pad_amount   = 4 DWORDs (16B)
    g1[1] = (int)((H & 0xFFFFu) << 16);                      // tensor_dim0[15:0] @bit48
    g1[2] = (int)((H >> 16) | ((BATCH & 0xFFFFu) << 16));    // dim0[31:16] | dim1[15:0]
    g1[3] = (int)((BATCH >> 16) | ((unsigned)KCH << 16));    // dim1[31:16] | tile_dim0
    g1[4] = 32;                                              // tile_dim1=32, tile_dim2=0
    g1[5] = H;                                               // tensor_dim0_stride[31:0]
    g1[6] = 0;                                               // stride hi | dim1_stride lo
    g1[7] = 0;
    i32x4 gz4 = {0, 0, 0, 0};                                // groups 2/3 unused (2D)
    i32x8 gz8 = {0, 0, 0, 0, 0, 0, 0, 0};
    __builtin_amdgcn_tensor_load_to_lds(g0, g1, gz4, gz4, gz8, 0);
}

// ---------------- convert / pad f32 -> bf16 ----------------
__global__ void cvt_pad_kernel(const float* __restrict__ src, bf16_t* __restrict__ dst,
                               int rows, int srcK, int dstK) {
    int idx = blockIdx.x * blockDim.x + threadIdx.x;
    int total = rows * dstK;
    if (idx >= total) return;
    int r = idx / dstK;
    int k = idx - r * dstK;
    float v = (k < srcK) ? src[(size_t)r * srcK + k] : 0.0f;
    dst[idx] = (bf16_t)v;
}

// ---------------- init output with b3 ----------------
__global__ void init_out_kernel(float* __restrict__ out, const float* __restrict__ b3, int n) {
    int idx = blockIdx.x * blockDim.x + threadIdx.x;
    if (idx < n) out[idx] = b3[0];
}

// ---------------- h0 = lag @ W1^T + b1 ----------------
// M=512, N=1024, K=192 (padded). Block: 256 thr = 8 waves (2 M x 4 N); tile 32x64.
__global__ __launch_bounds__(256) void h0_kernel(
    const bf16_t* __restrict__ lag_bf,   // [512][192]
    const bf16_t* __restrict__ W1_bf,    // [1024][192]
    const float*  __restrict__ b1,       // [1024]
    float* __restrict__ hF, bf16_t* __restrict__ hBf) {

    const int lane = threadIdx.x & 31;
    const int w    = threadIdx.x >> 5;
    const int wm   = w >> 2, wn = w & 3;
    const int m0   = blockIdx.y * 32 + wm * 16;
    const int n0   = blockIdx.x * 64 + wn * 16;
    const int lm   = lane & 15;
    const int hi   = lane >> 4;
    const int mA   = m0 + lm;
    const int nB   = n0 + lm;
    const int kA   = hi * 8;
    const int kB   = hi * 16;

    f32x8 acc = {};
    const bf16_t* wrow = W1_bf + (size_t)nB * K1P;
    for (int k0 = 0; k0 < K1P; k0 += 32) {
        FragBF a, b;
        load_fragA(a, lag_bf, K1P, mA, k0 + kA);
        load_fragB(b, wrow, k0 + kB);
        acc = __builtin_amdgcn_wmma_f32_16x16x32_bf16(false, a.v, false, b.v,
                                                      (short)0, acc, false, false);
    }
    #pragma unroll
    for (int v = 0; v < 8; ++v) {
        int m = m0 + v + hi * 8;
        int n = n0 + lm;
        float h = acc[v] + b1[n];
        hF[(size_t)m * H + n]  = h;
        hBf[(size_t)m * H + n] = (bf16_t)h;
    }
}

// ---------------- one GRU step, fused gates + y head ----------------
// gh = h@W_hh^T, gi = x@W_ih^T ; r/z summed in one acc; n-gate kept split.
// Block: 256 thr = 8 waves (2 M x 4 N), tile 32(M) x 64(N of H), grid (16,16).
// A-tile of h: TDM tensor_load_to_lds, double-buffered, pipelined on TENSORcnt.
// x-tile: GLOBAL_LOAD_ASYNC_TO_LDS_B128 issued at entry, waited after the
// recurrent loop (fully overlapped with the 8 recurrent chunks).
__global__ __launch_bounds__(256) void gru_step_kernel(
    const bf16_t* __restrict__ hp_bf,   // [512][1024] previous h, bf16
    const float*  __restrict__ hp_f,    // [512][1024] previous h, f32
    const bf16_t* __restrict__ x_bf,    // [512][64]   this step's input
    const bf16_t* __restrict__ Whh,     // [3072][1024]
    const bf16_t* __restrict__ Wih,     // [3072][64]
    const float*  __restrict__ b_ih,    // [3072]
    const float*  __restrict__ b_hh,    // [3072]
    const float*  __restrict__ W3,      // [1024]
    float* __restrict__ hn_f, bf16_t* __restrict__ hn_bf,
    float* __restrict__ out,            // [B][T], atomically accumulated
    int t) {

    __shared__ __align__(16) bf16_t smA[2][32][AROW];
    __shared__ __align__(16) bf16_t smX[32][XROW];

    const int tid  = threadIdx.x;
    const int lane = tid & 31;
    const int w    = tid >> 5;
    const int wm   = w >> 2, wn = w & 3;
    const int mBlk = blockIdx.y * 32;
    const int m0   = mBlk + wm * 16;
    const int n0   = blockIdx.x * 64 + wn * 16;
    const int lm   = lane & 15;
    const int hi   = lane >> 4;
    const int nB   = n0 + lm;
    const int kA   = hi * 8;
    const int kB   = hi * 16;
    const int lrow = wm * 16 + lm;      // local A-tile row this lane's fragment uses

    // LDS aperture lives in addr[63:32]; addr[31:0] is the LDS byte offset.
    const unsigned int smA_base = (unsigned int)(uintptr_t)(void*)&smA[0][0][0];
    const unsigned int smX_base = (unsigned int)(uintptr_t)(void*)&smX[0][0];

    // ---- kick off async x-tile prefetch: 256 x 16B segments, 1 per thread ----
    {
        int row = tid >> 3;          // 32 rows
        int c8  = tid & 7;           // 8 x 16B per 128B row
        unsigned long long gptr =
            (unsigned long long)(uintptr_t)(x_bf + (size_t)(mBlk + row) * INF_ + c8 * 8);
        unsigned int loff = smX_base + (unsigned int)row * (XROW * 2u)
                          + (unsigned int)c8 * 16u;
        asm volatile("global_load_async_to_lds_b128 %0, %1, off"
                     :: "v"(loff), "v"(gptr) : "memory");
    }

    f32x8 acc_r = {}, acc_z = {}, acc_gin = {}, acc_ghn = {};

    const bf16_t* row_r = Whh + (size_t)(nB)         * H;
    const bf16_t* row_z = Whh + (size_t)(H + nB)     * H;
    const bf16_t* row_n = Whh + (size_t)(2 * H + nB) * H;

    // ---- recurrent part: K = 1024 over previous h, TDM double-buffered ----
    const bf16_t* hRow = hp_bf + (size_t)mBlk * H;
    if (w == 0) {
        tdm_load_h_tile(hRow, smA_base);                       // chunk 0 -> buf 0
    }
    for (int c = 0; c < NCH; ++c) {
        const int buf = c & 1;
        if (w == 0) {
            if (c + 1 < NCH) {
                tdm_load_h_tile(hRow + (c + 1) * KCH,
                                smA_base + (unsigned int)(buf ^ 1) * (32u * AROW * 2u));
                // TDM ops from one wave complete in order: <=1 outstanding
                // means chunk c has fully landed in LDS.
                __builtin_amdgcn_s_wait_tensorcnt((short)1);
            } else {
                __builtin_amdgcn_s_wait_tensorcnt((short)0);
            }
        }
        __syncthreads();   // chunk c visible to every wave

        const int kc = c * KCH;
        #pragma unroll
        for (int kk = 0; kk < KCH; kk += 32) {
            FragBF a, br, bz, bn;
            const bf16_t* ap = &smA[buf][lrow][kk + kA];
            a.q[0] = *(const uint4*)(ap);
            a.q[1] = *(const uint4*)(ap + 16);
            load_fragB(br, row_r, kc + kk + kB);
            load_fragB(bz, row_z, kc + kk + kB);
            load_fragB(bn, row_n, kc + kk + kB);
            acc_r   = __builtin_amdgcn_wmma_f32_16x16x32_bf16(false, a.v, false, br.v,
                                                              (short)0, acc_r,   false, false);
            acc_z   = __builtin_amdgcn_wmma_f32_16x16x32_bf16(false, a.v, false, bz.v,
                                                              (short)0, acc_z,   false, false);
            acc_ghn = __builtin_amdgcn_wmma_f32_16x16x32_bf16(false, a.v, false, bn.v,
                                                              (short)0, acc_ghn, false, false);
        }
        __syncthreads();   // done reading smA[buf]; next TDM may overwrite it
    }

    // ---- input part: K = 64 over x_t from the async-prefetched LDS tile ----
    asm volatile("s_wait_asynccnt 0x0" ::: "memory");
    __syncthreads();       // every wave's x segments are in LDS

    const bf16_t* irow_r = Wih + (size_t)(nB)         * INF_;
    const bf16_t* irow_z = Wih + (size_t)(H + nB)     * INF_;
    const bf16_t* irow_n = Wih + (size_t)(2 * H + nB) * INF_;
    #pragma unroll
    for (int k0 = 0; k0 < INF_; k0 += 32) {
        FragBF a, br, bz, bn;
        const bf16_t* ap = &smX[lrow][k0 + kA];
        a.q[0] = *(const uint4*)(ap);
        a.q[1] = *(const uint4*)(ap + 16);
        load_fragB(br, irow_r, k0 + kB);
        load_fragB(bz, irow_z, k0 + kB);
        load_fragB(bn, irow_n, k0 + kB);
        acc_r   = __builtin_amdgcn_wmma_f32_16x16x32_bf16(false, a.v, false, br.v,
                                                          (short)0, acc_r,   false, false);
        acc_z   = __builtin_amdgcn_wmma_f32_16x16x32_bf16(false, a.v, false, bz.v,
                                                          (short)0, acc_z,   false, false);
        acc_gin = __builtin_amdgcn_wmma_f32_16x16x32_bf16(false, a.v, false, bn.v,
                                                          (short)0, acc_gin, false, false);
    }

    // ---- gate math + h update + fused y partial ----
    const int n = n0 + lm;
    const float bir = b_ih[n]         + b_hh[n];
    const float biz = b_ih[H + n]     + b_hh[H + n];
    const float bin = b_ih[2 * H + n];
    const float bhn = b_hh[2 * H + n];
    const float w3  = W3[n];

    float yp[8];
    #pragma unroll
    for (int v = 0; v < 8; ++v) {
        int m = m0 + v + hi * 8;
        float r  = sigmoid_f(acc_r[v] + bir);
        float z  = sigmoid_f(acc_z[v] + biz);
        float nn = tanh_f(acc_gin[v] + bin + r * (acc_ghn[v] + bhn));
        float ho = hp_f[(size_t)m * H + n];
        float hn = (1.0f - z) * nn + z * ho;
        hn_f [(size_t)m * H + n] = hn;
        hn_bf[(size_t)m * H + n] = (bf16_t)hn;
        yp[v] = hn * w3;
    }

    // reduce y partials across the 16 lanes sharing each M-row, atomically add
    #pragma unroll
    for (int v = 0; v < 8; ++v) {
        float s = yp[v];
        s += __shfl_xor(s, 1, 16);
        s += __shfl_xor(s, 2, 16);
        s += __shfl_xor(s, 4, 16);
        s += __shfl_xor(s, 8, 16);
        if (lm == 0) {
            int m = m0 + v + hi * 8;
            atomicAdd(&out[(size_t)m * TT + t], s);
        }
    }
}

// ---------------- host launcher ----------------
extern "C" void kernel_launch(void* const* d_in, const int* in_sizes, int n_in,
                              void* d_out, int out_size, void* d_ws, size_t ws_size,
                              hipStream_t stream) {
    (void)in_sizes; (void)n_in; (void)out_size; (void)ws_size;

    const float* lag  = (const float*)d_in[0];  // [B][LAG]
    const float* curr = (const float*)d_in[1];  // [T][B][IN]
    const float* W1   = (const float*)d_in[2];  // [H][LAG]
    const float* b1   = (const float*)d_in[3];  // [H]
    const float* W_ih = (const float*)d_in[4];  // [3H][IN]
    const float* b_ih = (const float*)d_in[5];  // [3H]
    const float* W_hh = (const float*)d_in[6];  // [3H][H]
    const float* b_hh = (const float*)d_in[7];  // [3H]
    const float* W3   = (const float*)d_in[8];  // [1][H]
    const float* b3   = (const float*)d_in[9];  // [1]
    float* out = (float*)d_out;                 // [B][T]

    char* ws = (char*)d_ws;
    size_t off = 0;
    auto wsalloc = [&](size_t bytes) -> char* {
        char* p = ws + off;
        off += (bytes + 255) & ~(size_t)255;
        return p;
    };
    bf16_t* Whh_bf  = (bf16_t*)wsalloc((size_t)H3 * H * 2);
    bf16_t* Wih_bf  = (bf16_t*)wsalloc((size_t)H3 * INF_ * 2);
    bf16_t* W1_bf   = (bf16_t*)wsalloc((size_t)H * K1P * 2);
    bf16_t* lag_bf  = (bf16_t*)wsalloc((size_t)BATCH * K1P * 2);
    bf16_t* curr_bf = (bf16_t*)wsalloc((size_t)TT * BATCH * INF_ * 2);
    float*  hA_f    = (float*) wsalloc((size_t)BATCH * H * 4);
    float*  hB_f    = (float*) wsalloc((size_t)BATCH * H * 4);
    bf16_t* hA_bf   = (bf16_t*)wsalloc((size_t)BATCH * H * 2);
    bf16_t* hB_bf   = (bf16_t*)wsalloc((size_t)BATCH * H * 2);

    auto launch_cvt = [&](const float* src, bf16_t* dst, int rows, int srcK, int dstK) {
        int total = rows * dstK;
        cvt_pad_kernel<<<(total + 255) / 256, 256, 0, stream>>>(src, dst, rows, srcK, dstK);
    };
    launch_cvt(W_hh, Whh_bf, H3, H, H);
    launch_cvt(W_ih, Wih_bf, H3, INF_, INF_);
    launch_cvt(W1,   W1_bf,  H,  LAG, K1P);
    launch_cvt(lag,  lag_bf, BATCH, LAG, K1P);
    launch_cvt(curr, curr_bf, TT * BATCH, INF_, INF_);

    init_out_kernel<<<(BATCH * TT + 255) / 256, 256, 0, stream>>>(out, b3, BATCH * TT);

    dim3 gemm_grid(H / 64, BATCH / 32);
    h0_kernel<<<gemm_grid, 256, 0, stream>>>(lag_bf, W1_bf, b1, hA_f, hA_bf);

    for (int t = 0; t < TT; ++t) {
        const bf16_t* hp_bf = (t & 1) ? hB_bf : hA_bf;
        const float*  hp_f  = (t & 1) ? hB_f  : hA_f;
        float*        hn_f  = (t & 1) ? hA_f  : hB_f;
        bf16_t*       hn_bf = (t & 1) ? hA_bf : hB_bf;
        const bf16_t* x_bf  = curr_bf + (size_t)t * BATCH * INF_;
        gru_step_kernel<<<gemm_grid, 256, 0, stream>>>(
            hp_bf, hp_f, x_bf, Whh_bf, Wih_bf, b_ih, b_hh, W3,
            hn_f, hn_bf, out, t);
    }
}
